// Memory_Bank_Att_10118942949622
// MI455X (gfx1250) — matrix operand order repaired and checked by reading, not compiled, not run on hardware
//
#include <hip/hip_runtime.h>
#include <hip/hip_bf16.h>

// ---------------------------------------------------------------------------
// Problem constants (from reference setup_inputs)
// ---------------------------------------------------------------------------
constexpr int B   = 16;
constexpr int N   = 8192;
constexpr int C   = 128;   // dim
constexpr int H   = 2;     // heads
constexpr int TILE = 128;  // positions per workgroup in the partials pass
constexpr int NB  = N / TILE;  // 64 tiles per batch

typedef __attribute__((ext_vector_type(8)))  float  v8f;
typedef __attribute__((ext_vector_type(4)))  __bf16 v4bf;
typedef __attribute__((ext_vector_type(8)))  __bf16 v8bf;
typedef __attribute__((ext_vector_type(16))) __bf16 v16bf;

// ---------------------------------------------------------------------------
// Kernel 0: wkeff[h, g*64+i] = sum_j mem[h, 2j+g] * Wk[g, h*32+j, i]
// Collapses mem . (grouped-linear k) into a single 128-vector per head.
// ---------------------------------------------------------------------------
__global__ void wkeff_kernel(const float* __restrict__ Wk,
                             const float* __restrict__ mem,
                             float* __restrict__ wkeff) {
    int t = threadIdx.x;            // 256 threads: t = h*128 + c
    int h = t >> 7;
    int c = t & 127;
    int g = c >> 6;
    int i = c & 63;
    float acc = 0.0f;
#pragma unroll
    for (int j = 0; j < 32; ++j) {
        acc += mem[h * 64 + 2 * j + g] * Wk[g * 4096 + (h * 32 + j) * 64 + i];
    }
    wkeff[t] = acc;
}

// ---------------------------------------------------------------------------
// Kernel 1: one pass over x. Per (b, tile of 128 positions):
//   - stage x tile in LDS as bf16 (row-major, pos x chan)
//   - logits s[pos,h] = x . (scale*wkeff) via v_wmma_f32_16x16x32_bf16:
//       A = x rows, B = wk_n[16][128] (zero-padded head columns), both as
//       contiguous aligned ds_load_b128 pairs. One 16-pos M-tile per wave.
//   - softmax partials via wave32 __shfl_xor butterflies (no barrier trees)
//   - xbar_partial[h][c] = sum_pos w[h][pos]*x[pos][c], exact f32 per-thread
// ---------------------------------------------------------------------------
__global__ void partials_kernel(const float* __restrict__ x,
                                const float* __restrict__ wkeff,
                                float* __restrict__ pm,
                                float* __restrict__ pz,
                                float* __restrict__ pxbar) {
    const int b    = blockIdx.y;
    const int tile = blockIdx.x;
    const int n0   = tile * TILE;
    const int t    = threadIdx.x;          // 256 threads = 8 waves (wave32)

    __shared__ __align__(16) __bf16 xb[TILE][C];   // 32 KB bf16 x tile
    __shared__ __align__(16) __bf16 wk_n[16][C];   // 4 KB, rows >= H are zero
    __shared__ __align__(16) float  s_lds[H][TILE];
    __shared__ float mred[8];   // per-wave max partials
    __shared__ float zred[8];   // per-wave sum partials

    // --- stage wk_n = 0.125 * wkeff for rows 0..1, zero rows 2..15 ---
    for (int i = t; i < 16 * C; i += 256) {
        int hh = i >> 7, cc = i & 127;
        wk_n[hh][cc] = (hh < H) ? (__bf16)(0.125f * wkeff[hh * C + cc])
                                : (__bf16)0.0f;
    }

    // --- stage x tile: coalesced float4 global loads, packed b64 LDS stores ---
    const float4* xsrc = (const float4*)(x + ((size_t)b * N + n0) * C);
    for (int i = t; i < TILE * C / 4; i += 256) {
        float4 v = xsrc[i];
        int r  = i >> 5;        // C/4 = 32 float4 per row
        int c4 = (i & 31) * 4;
        v4bf pk = { (__bf16)v.x, (__bf16)v.y, (__bf16)v.z, (__bf16)v.w };
        *(v4bf*)&xb[r][c4] = pk;
    }
    __syncthreads();   // #1

    // --- logits via WMMA: wave wv owns positions [wv*16, wv*16+16) ---
    const int lane = t & 31;
    const int wv   = t >> 5;        // 0..7
    const int lrow = lane & 15;     // A row (pos) / B,D column (head)
    const int lhi  = lane >> 4;

    v8f sacc = {};
#pragma unroll
    for (int ks = 0; ks < C / 32; ++ks) {   // 4 K-steps of 32 channels
        // A (16-bit 16x32): lane<16 e=0..7 -> K=e, e=8..15 -> K=e+8; lane>=16 +8
        v8bf a0 = *(const v8bf*)&xb[wv * 16 + lrow][ks * 32 + lhi * 8];
        v8bf a1 = *(const v8bf*)&xb[wv * 16 + lrow][ks * 32 + 16 + lhi * 8];
        v16bf af = __builtin_shufflevector(a0, a1,
            0,1,2,3,4,5,6,7,8,9,10,11,12,13,14,15);
        // B (16-bit 32x16): same K order as A (e=0..7 -> K=e, e=8..15 -> K=e+8; lane>=16 +8)
        v8bf b0 = *(const v8bf*)&wk_n[lrow][ks * 32 + lhi * 8];
        v8bf b1 = *(const v8bf*)&wk_n[lrow][ks * 32 + 16 + lhi * 8];
        v16bf bf_ = __builtin_shufflevector(b0, b1,
            0,1,2,3,4,5,6,7,8,9,10,11,12,13,14,15);
        sacc = __builtin_amdgcn_wmma_f32_16x16x32_bf16(
            false, af, false, bf_, (short)0, sacc, false, false);
    }

    // D (32-bit 16x16): element r -> pos = wv*16 + lhi*8 + r, column = head.
    if (lrow < H) {
        float4 lo = make_float4(sacc[0], sacc[1], sacc[2], sacc[3]);
        float4 hi = make_float4(sacc[4], sacc[5], sacc[6], sacc[7]);
        *(float4*)&s_lds[lrow][wv * 16 + lhi * 8]     = lo;
        *(float4*)&s_lds[lrow][wv * 16 + lhi * 8 + 4] = hi;
    }
    __syncthreads();   // #2

    // --- softmax partials: waves 0-3 -> head 0, waves 4-7 -> head 1 ---
    const int pos = t & 127;
    const int h   = t >> 7;
    const float s = s_lds[h][pos];

    // wave max via shfl_xor butterfly (wave32, no barriers)
    float m = s;
#pragma unroll
    for (int off = 16; off; off >>= 1)
        m = fmaxf(m, __shfl_xor(m, off, 32));
    if (lane == 0) mred[wv] = m;
    __syncthreads();   // #3

    // per-head max = max of this head's 4 wave partials (broadcast reads)
    const float mh = fmaxf(fmaxf(mred[h * 4 + 0], mred[h * 4 + 1]),
                           fmaxf(mred[h * 4 + 2], mred[h * 4 + 3]));
    const float w = __expf(s - mh);
    s_lds[h][pos] = w;                 // overwrite logits with exp weights

    // wave sum via shfl_xor butterfly
    float zs = w;
#pragma unroll
    for (int off = 16; off; off >>= 1)
        zs += __shfl_xor(zs, off, 32);
    if (lane == 0) zred[wv] = zs;
    __syncthreads();   // #4 (also publishes s_lds weights)

    const int pidx = (b * NB + tile) * H;
    if (t == 0) {
        pm[pidx] = mh;
        pz[pidx] = zred[0] + zred[1] + zred[2] + zred[3];
    }
    if (t == 128) {
        pm[pidx + 1] = mh;
        pz[pidx + 1] = zred[4] + zred[5] + zred[6] + zred[7];
    }

    // --- xbar partial: thread t = (head h, channel c), exact f32 ---
    // s_lds[h][p] is a broadcast; xb[p][c] is a conflict-free row read.
    const int c = pos;
    float acc = 0.0f;
#pragma unroll 8
    for (int p = 0; p < TILE; ++p) {
        acc += s_lds[h][p] * (float)xb[p][c];
    }
    pxbar[((size_t)(b * NB + tile) * H + h) * C + c] = acc;
}

// ---------------------------------------------------------------------------
// Kernel 2: per batch b (16 blocks x 128 threads):
//   combine tile partials -> global softmax -> xbar -> row = Wv*xbar
//   -> out_pre -> final = Wp*out_pre + bp   (128 floats per batch)
// ---------------------------------------------------------------------------
__global__ void finalize_kernel(const float* __restrict__ Wv,
                                const float* __restrict__ Wp,
                                const float* __restrict__ bp,
                                const float* __restrict__ pm,
                                const float* __restrict__ pz,
                                const float* __restrict__ pxbar,
                                float* __restrict__ finalv) {
    const int b = blockIdx.x;
    const int t = threadIdx.x;   // 128 threads

    __shared__ float Msh[2], Zsh[2];
    __shared__ float xbar_s[H][C];
    __shared__ float outpre[C];

    if (t < 2) {
        float m = -1e30f;
        for (int tl = 0; tl < NB; ++tl)
            m = fmaxf(m, pm[(b * NB + tl) * H + t]);
        float z = 0.0f;
        for (int tl = 0; tl < NB; ++tl) {
            int pi = (b * NB + tl) * H + t;
            z += pz[pi] * __expf(pm[pi] - m);
        }
        Msh[t] = m;
        Zsh[t] = z;
    }
    __syncthreads();

    // xbar[h][c], thread t = channel c
    for (int h = 0; h < H; ++h) {
        float acc = 0.0f;
        for (int tl = 0; tl < NB; ++tl) {
            int pi = (b * NB + tl) * H + h;
            acc += pxbar[(size_t)pi * C + t] * __expf(pm[pi] - Msh[h]);
        }
        xbar_s[h][t] = acc / Zsh[h];
    }
    __syncthreads();

    // row[h][d] = sum_i Wv[g, h*32 + d/2, i] * xbar[h, g*64+i], g = d&1
    {
        int h = t >> 6, d = t & 63;
        int g = d & 1, o = h * 32 + (d >> 1);
        float acc = 0.0f;
#pragma unroll 8
        for (int i = 0; i < 64; ++i)
            acc += Wv[g * 4096 + o * 64 + i] * xbar_s[h][g * 64 + i];
        outpre[h * 64 + d] = acc;   // out_pre[c = h*64+d]
    }
    __syncthreads();

    // final[c' = o*2+g] = bp[g,o] + sum_i Wp[g,o,i] * out_pre[g*64+i]
    {
        int g = t & 1, o = t >> 1;
        float acc = bp[g * 64 + o];
#pragma unroll 8
        for (int i = 0; i < 64; ++i)
            acc += Wp[g * 4096 + o * 64 + i] * outpre[g * 64 + i];
        finalv[b * C + t] = acc;
    }
}

// ---------------------------------------------------------------------------
// Kernel 3: broadcast final row to every position: out[b,n,:] = final[b,:]
// Pure 67 MB float4 stream — this + the x read in kernel 1 are the roofline.
// ---------------------------------------------------------------------------
__global__ void broadcast_kernel(const float* __restrict__ finalv,
                                 float4* __restrict__ out) {
    size_t q = (size_t)blockIdx.x * blockDim.x + threadIdx.x;  // float4 index
    int b  = (int)(q >> 18);      // N*C/4 = 262144 = 2^18 float4 per batch
    int c4 = (int)(q & 31);       // C/4 = 32
    const float4* f4 = (const float4*)finalv;
    out[q] = f4[b * 32 + c4];
}

// ---------------------------------------------------------------------------
// Host launcher
// ---------------------------------------------------------------------------
extern "C" void kernel_launch(void* const* d_in, const int* in_sizes, int n_in,
                              void* d_out, int out_size, void* d_ws, size_t ws_size,
                              hipStream_t stream) {
    const float* x   = (const float*)d_in[0];
    const float* Wk  = (const float*)d_in[1];
    const float* Wv  = (const float*)d_in[2];
    const float* Wp  = (const float*)d_in[3];
    const float* bp  = (const float*)d_in[4];
    const float* mem = (const float*)d_in[5];
    float* out = (float*)d_out;

    // Workspace layout (floats)
    float* ws      = (float*)d_ws;
    float* wkeff   = ws;                       // 256
    float* pm      = wkeff + 256;              // B*NB*H = 2048
    float* pz      = pm + B * NB * H;          // 2048
    float* pxbar   = pz + B * NB * H;          // B*NB*H*C = 262144
    float* finalv  = pxbar + (size_t)B * NB * H * C;  // B*C = 2048

    wkeff_kernel<<<1, 256, 0, stream>>>(Wk, mem, wkeff);
    partials_kernel<<<dim3(NB, B), 256, 0, stream>>>(x, wkeff, pm, pz, pxbar);
    finalize_kernel<<<B, 128, 0, stream>>>(Wv, Wp, bp, pm, pz, pxbar, finalv);

    const int tot4 = B * N * C / 4;            // 4,194,304 float4
    broadcast_kernel<<<tot4 / 256, 256, 0, stream>>>(finalv, (float4*)out);
}
